// MIMoE_75076028334264
// MI455X (gfx1250) — matrix operand order repaired and checked by reading, not compiled, not run on hardware
//
#include <hip/hip_runtime.h>
#include <hip/hip_bf16.h>

#define N_TOK 8192
#define HID   1024
#define NEXP  16
#define TOPK  4
#define ALPHA_LOSS 0.001f

#define XST  1032   // LDS row stride in bf16 elems (1024 + 8 pad -> conflict-free frag loads)
#define MTOK 32     // tokens per workgroup (two 16-row M tiles)

typedef __bf16 bf16_t;
typedef __attribute__((ext_vector_type(16))) __bf16 v16bf;
typedef __attribute__((ext_vector_type(8)))  __bf16 v8bf;
typedef __attribute__((ext_vector_type(4)))  __bf16 v4bf;
typedef __attribute__((ext_vector_type(8)))  float  v8f;
typedef __attribute__((ext_vector_type(4)))  float  v4f;

__device__ __forceinline__ v8f wmma_bf16(v16bf a, v16bf b, v8f c) {
  // D(f32 16x16) = A(bf16 16x32) * B(bf16 32x16) + C
  return __builtin_amdgcn_wmma_f32_16x16x32_bf16(false, a, false, b, (short)0, c, false, false);
}

// A fragment (16x32 bf16, rows = tokens) from an LDS tile.
// A layout (matches B): lanes 0-15 row=lane, elems0-15 -> K 0-15;
// lanes 16-31 same rows, K 16-31.
__device__ __forceinline__ v16bf load_a_lds(const bf16_t* base, int kc, int lane) {
  const int row = lane & 15;
  const int klo = kc * 32 + ((lane & 16) ? 16 : 0);
  const bf16_t* p = base + row * XST + klo;
  v8bf lo = *(const v8bf*)(p);
  v8bf hi = *(const v8bf*)(p + 8);
  return __builtin_shufflevector(lo, hi, 0,1,2,3,4,5,6,7,8,9,10,11,12,13,14,15);
}

// ---- B fragment loaders -------------------------------------------------
// B layout (32x16): lanes 0-15 col=lane hold K 0-15 sequentially; lanes 16-31 hold K 16-31.

// Fallback: from global f32 row-major [out,in] weights, convert in-loop.
__device__ __forceinline__ v16bf load_b_f32(const float* wbase, int kc, int lane) {
  const int col = lane & 15;
  const int klo = kc * 32 + ((lane & 16) ? 16 : 0);
  const float* p = wbase + (size_t)col * HID + klo;
  v4f f0 = *(const v4f*)(p);
  v4f f1 = *(const v4f*)(p + 4);
  v4f f2 = *(const v4f*)(p + 8);
  v4f f3 = *(const v4f*)(p + 12);
  v16bf r;
#pragma unroll
  for (int j = 0; j < 4; ++j) {
    r[j]      = (bf16_t)f0[j];
    r[4 + j]  = (bf16_t)f1[j];
    r[8 + j]  = (bf16_t)f2[j];
    r[12 + j] = (bf16_t)f3[j];
  }
  return r;
}

// Fast path: pre-swizzled bf16 weights, layout [tile][kc][lane][16 halves]
// -> whole wave reads one contiguous 1KB burst per fragment.
__device__ __forceinline__ v16bf load_b_swz(const bf16_t* tile_base, int kc, int lane) {
  return *(const v16bf*)(tile_base + (size_t)kc * 512 + (size_t)lane * 16);
}

// ---- small kernels ------------------------------------------------------

__global__ void moe_init(int* ws) {
  if (threadIdx.x < 48) ws[threadIdx.x] = 0;
}

__global__ void moe_topk(const float* __restrict__ scores, int* __restrict__ topk,
                         int* __restrict__ counts) {
  const int n = blockIdx.x * blockDim.x + threadIdx.x;
  if (n >= N_TOK) return;
  float s[NEXP];
#pragma unroll
  for (int e = 0; e < NEXP; ++e) s[e] = scores[n * NEXP + e];
#pragma unroll
  for (int slot = 0; slot < TOPK; ++slot) {
    float best = s[0];
    int bi = 0;
#pragma unroll
    for (int e = 1; e < NEXP; ++e)
      if (s[e] > best) { best = s[e]; bi = e; }   // strict > : lowest index wins ties (jax top_k)
    topk[n * TOPK + slot] = bi;
    s[bi] = -__builtin_inff();
    atomicAdd(&counts[bi], 1);
  }
}

__global__ void moe_reduce(const float* __restrict__ scores, const int* __restrict__ counts,
                           int* __restrict__ offsets, int* __restrict__ cursors,
                           float* __restrict__ aux) {
  __shared__ float part[256 * NEXP];
  float loc[NEXP];
#pragma unroll
  for (int e = 0; e < NEXP; ++e) loc[e] = 0.f;
  for (int n = threadIdx.x; n < N_TOK; n += 256) {
#pragma unroll
    for (int e = 0; e < NEXP; ++e) loc[e] += scores[n * NEXP + e];
  }
#pragma unroll
  for (int e = 0; e < NEXP; ++e) part[threadIdx.x * NEXP + e] = loc[e];
  __syncthreads();
  for (int st = 128; st > 0; st >>= 1) {   // fixed-order tree -> deterministic
    if ((int)threadIdx.x < st)
      for (int e = 0; e < NEXP; ++e)
        part[threadIdx.x * NEXP + e] += part[(threadIdx.x + st) * NEXP + e];
    __syncthreads();
  }
  if (threadIdx.x == 0) {
    int off = 0;
    float a = 0.f;
    for (int e = 0; e < NEXP; ++e) {
      offsets[e] = off;
      cursors[e] = off;
      off += counts[e];
      float Pi = part[e] / (float)N_TOK;
      float ce = (float)counts[e] / (float)(N_TOK * TOPK);
      a += Pi * ce * (float)NEXP;
    }
    aux[0] = a * ALPHA_LOSS;
  }
}

__global__ void moe_scatter(const int* __restrict__ topk, int* __restrict__ cursors,
                            int* __restrict__ pairs) {
  const int i = blockIdx.x * blockDim.x + threadIdx.x;
  if (i >= N_TOK * TOPK) return;
  const int e = topk[i];
  const int pos = atomicAdd(&cursors[e], 1);
  pairs[pos] = i;            // i encodes token*4 + slot
}

// f32 -> bf16 weight conversion + swizzle to B-fragment order.
// wswz layout: [mat(3)][expert(16)] blocks of 1M halves; within a block:
//   [tile(64)][kc(32)][lane(32)][16 halves]
__global__ void moe_convert(const float* __restrict__ Wg, const float* __restrict__ Wu,
                            const float* __restrict__ Wd, bf16_t* __restrict__ wswz) {
  const int id = blockIdx.x * 256 + threadIdx.x;  // 3*16*65536 threads
  const int g  = id & 65535;
  const int me = id >> 16;                        // mat*16 + expert
  const int mat = me >> 4;
  const int lane = g & 31;
  const int kc   = (g >> 5) & 31;
  const int it   = g >> 10;
  const float* W = (mat == 0 ? Wg : (mat == 1 ? Wu : Wd)) + ((size_t)(me & 15) << 20);
  const int row = it * 16 + (lane & 15);
  const int k0  = kc * 32 + ((lane & 16) ? 16 : 0);
  const float* src = W + (size_t)row * HID + k0;
  v4f f0 = *(const v4f*)(src);
  v4f f1 = *(const v4f*)(src + 4);
  v4f f2 = *(const v4f*)(src + 8);
  v4f f3 = *(const v4f*)(src + 12);
  v16bf r;
#pragma unroll
  for (int j = 0; j < 4; ++j) {
    r[j]      = (bf16_t)f0[j];
    r[4 + j]  = (bf16_t)f1[j];
    r[8 + j]  = (bf16_t)f2[j];
    r[12 + j] = (bf16_t)f3[j];
  }
  *(v16bf*)(wswz + ((size_t)me << 20) + (size_t)g * 16) = r;
}

// ---- main fused grouped-GEMM MLP ---------------------------------------

template <bool PRE>
__global__ void __launch_bounds__(256)
moe_mlp(const float* __restrict__ hs, const float* __restrict__ Wg,
        const float* __restrict__ Wu, const float* __restrict__ Wd,
        const bf16_t* __restrict__ wswz,
        const int* __restrict__ counts, const int* __restrict__ offsets,
        const int* __restrict__ pairs, float* __restrict__ out) {
  extern __shared__ char smem[];
  bf16_t* Xs = (bf16_t*)smem;                 // [MTOK][XST] bf16 token rows
  bf16_t* As = Xs + MTOK * XST;               // [MTOK][XST] bf16 silu(g)*u
  int* rowcode = (int*)(As + MTOK * XST);     // [MTOK] n*4+slot per tile row

  const int e = blockIdx.y;
  const int t = blockIdx.x;
  const int total = counts[e];
  const int base = t * MTOK;
  if (base >= total) return;                  // block-uniform exit
  int cnt = total - base;
  if (cnt > MTOK) cnt = MTOK;
  const int start = offsets[e] + base;

  const int tid  = threadIdx.x;
  const int lane = tid & 31;
  const int wave = tid >> 5;

  if (tid < MTOK) rowcode[tid] = (tid < cnt) ? pairs[start + tid] : -1;
  __syncthreads();

  // stage X (f32 -> bf16) into LDS; zero-pad unused rows
  for (int idx = tid; idx < MTOK * 256; idx += 256) {
    const int row = idx >> 8;
    const int c4  = (idx & 255) << 2;
    const int code = rowcode[row];
    bf16_t* dst = Xs + row * XST + c4;
    v4bf w;
    if (code >= 0) {
      const float* src = hs + (size_t)(code >> 2) * HID + c4;
      v4f v = *(const v4f*)src;
      w[0] = (bf16_t)v[0]; w[1] = (bf16_t)v[1];
      w[2] = (bf16_t)v[2]; w[3] = (bf16_t)v[3];
    } else {
      w[0] = (bf16_t)0.f; w[1] = (bf16_t)0.f; w[2] = (bf16_t)0.f; w[3] = (bf16_t)0.f;
    }
    *(v4bf*)dst = w;
  }
  __syncthreads();

  const float* Wge = Wg + ((size_t)e << 20);
  const float* Wue = Wu + ((size_t)e << 20);
  const float* Wde = Wd + ((size_t)e << 20);
  const bf16_t* Wgs = wswz + ((size_t)(0 * NEXP + e) << 20);
  const bf16_t* Wus = wswz + ((size_t)(1 * NEXP + e) << 20);
  const bf16_t* Wds = wswz + ((size_t)(2 * NEXP + e) << 20);

  // Phase 1: g = X*Wg^T, u = X*Wu^T ; act = silu(g)*u  -> LDS (bf16)
  for (int it = wave; it < 64; it += 8) {
    const int i0 = it * 16;
    v8f g0 = {}, g1 = {}, u0 = {}, u1 = {};
    const float*  wgf = Wge + (size_t)i0 * HID;
    const float*  wuf = Wue + (size_t)i0 * HID;
    const bf16_t* wgt = Wgs + (size_t)it * 16384;
    const bf16_t* wut = Wus + (size_t)it * 16384;
#pragma unroll 4
    for (int kc = 0; kc < 32; ++kc) {         // K = 1024 in chunks of 32
      v16bf a0 = load_a_lds(Xs, kc, lane);
      v16bf a1 = load_a_lds(Xs + 16 * XST, kc, lane);
      v16bf bg = PRE ? load_b_swz(wgt, kc, lane) : load_b_f32(wgf, kc, lane);
      v16bf bu = PRE ? load_b_swz(wut, kc, lane) : load_b_f32(wuf, kc, lane);
      g0 = wmma_bf16(a0, bg, g0);
      g1 = wmma_bf16(a1, bg, g1);
      u0 = wmma_bf16(a0, bu, u0);
      u1 = wmma_bf16(a1, bu, u1);
    }
    const int colr  = lane & 15;
    const int rbase = (lane >> 4) << 3;       // D layout: vgpr r -> row r (+8 for hi lanes)
#pragma unroll
    for (int r = 0; r < 8; ++r) {
      float g = g0[r], u = u0[r];
      float s = (g / (1.f + __expf(-g))) * u;
      As[(rbase + r) * XST + i0 + colr] = (bf16_t)s;
      g = g1[r]; u = u1[r];
      s = (g / (1.f + __expf(-g))) * u;
      As[(16 + rbase + r) * XST + i0 + colr] = (bf16_t)s;
    }
  }
  __syncthreads();

  // Phase 2: out = relu(act * Wd^T), scattered to d_out rows
  for (int ht = wave; ht < 64; ht += 8) {
    const int h0 = ht * 16;
    v8f c0 = {}, c1 = {};
    const float*  wdf = Wde + (size_t)h0 * HID;
    const bf16_t* wdt = Wds + (size_t)ht * 16384;
#pragma unroll 4
    for (int kc = 0; kc < 32; ++kc) {
      v16bf a0 = load_a_lds(As, kc, lane);
      v16bf a1 = load_a_lds(As + 16 * XST, kc, lane);
      v16bf bd = PRE ? load_b_swz(wdt, kc, lane) : load_b_f32(wdf, kc, lane);
      c0 = wmma_bf16(a0, bd, c0);
      c1 = wmma_bf16(a1, bd, c1);
    }
    const int colr  = lane & 15;
    const int rbase = (lane >> 4) << 3;
#pragma unroll
    for (int r = 0; r < 8; ++r) {
      int row = rbase + r;
      if (row < cnt)
        out[(size_t)rowcode[row] * HID + h0 + colr] = fmaxf(c0[r], 0.f);
      row += 16;
      if (row < cnt)
        out[(size_t)rowcode[row] * HID + h0 + colr] = fmaxf(c1[r], 0.f);
    }
  }
}

extern "C" void kernel_launch(void* const* d_in, const int* in_sizes, int n_in,
                              void* d_out, int out_size, void* d_ws, size_t ws_size,
                              hipStream_t stream) {
  const float* hs     = (const float*)d_in[0];
  const float* scores = (const float*)d_in[1];
  const float* Wg     = (const float*)d_in[2];
  const float* Wu     = (const float*)d_in[3];
  const float* Wd     = (const float*)d_in[4];
  float* out = (float*)d_out;

  // ws layout: routing ints in first 1MB, swizzled bf16 weights after (96MB).
  int* ws      = (int*)d_ws;
  int* counts  = ws;                       // [16]
  int* offsets = ws + 16;                  // [16]
  int* cursors = ws + 32;                  // [16]
  int* topk    = ws + 64;                  // [N*K]
  int* pairs   = ws + 64 + N_TOK * TOPK;   // [N*K]
  bf16_t* wswz = (bf16_t*)((char*)d_ws + (1ull << 20));

  const size_t wbytes = (size_t)3 * NEXP * HID * HID * sizeof(bf16_t);  // 96 MB
  const bool pre = ws_size >= (1ull << 20) + wbytes;

  moe_init<<<1, 64, 0, stream>>>(ws);
  moe_topk<<<N_TOK / 256, 256, 0, stream>>>(scores, topk, counts);
  moe_reduce<<<1, 256, 0, stream>>>(scores, counts, offsets, cursors,
                                    out + (size_t)(out_size - 1));
  moe_scatter<<<(N_TOK * TOPK) / 256, 256, 0, stream>>>(topk, cursors, pairs);

  dim3 grid(256, NEXP);  // 256 = ceil(N / MTOK) worst case per expert; empty tiles exit fast
  const size_t smem = (size_t)2 * MTOK * XST * sizeof(bf16_t) + MTOK * sizeof(int);
  if (pre) {
    moe_convert<<<(3 * NEXP * 65536) / 256, 256, 0, stream>>>(Wg, Wu, Wd, wswz);
    moe_mlp<true><<<grid, 256, smem, stream>>>(hs, Wg, Wu, Wd, wswz,
                                               counts, offsets, pairs, out);
  } else {
    moe_mlp<false><<<grid, 256, smem, stream>>>(hs, Wg, Wu, Wd, wswz,
                                                counts, offsets, pairs, out);
  }
}
